// UpdateRule_9921374454398
// MI455X (gfx1250) — matrix operand order repaired and checked by reading, not compiled, not run on hardware
//
#include <hip/hip_runtime.h>

#define NN 32768
#define DEG 8

typedef __attribute__((ext_vector_type(16))) __bf16 v16bf;
typedef __attribute__((ext_vector_type(8)))  __bf16 v8bf;
typedef __attribute__((ext_vector_type(8)))  float  v8f;
typedef __attribute__((address_space(3)))    char   as3char;

// ---------------------------------------------------------------------------
// WMMA fragment helpers (bf16 16x16x32, wave32; A/B in contiguous k order, C/D per CDNA5 ISA 7.12.2)
// A (16x32, 16-bit): lane row r = lane&15, element e -> K = 16*h + e
//   => one contiguous 16-element run at kb+16h (loaded as two 16B halves).
// B (32x16, 16-bit): lane col n = lane&15, element e -> K = 16*h + e
//   => one contiguous 16-element (32B) run per lane from a [n][k] buffer.
// C/D (16x16 f32): lane col n = lane&15, VGPR i -> row M = i + 8*h.
// ---------------------------------------------------------------------------
__device__ __forceinline__ v16bf load_a_frag(const __bf16* buf, int ld, int row,
                                             int kb, int h) {
  const __bf16* p0 = buf + row * ld + kb + h * 16;
  v8bf lo = *(const v8bf*)p0;
  v8bf hi = *(const v8bf*)(p0 + 8);
  v16bf r;
#pragma unroll
  for (int i = 0; i < 8; ++i) { r[i] = lo[i]; r[i + 8] = hi[i]; }
  return r;
}

__device__ __forceinline__ v16bf load_b_frag(const __bf16* buf, int ld, int ncol,
                                             int kb, int h) {
  return *(const v16bf*)(buf + ncol * ld + kb + h * 16);
}

__device__ __forceinline__ v8f wmma_bf16(v16bf a, v16bf b, v8f c) {
  return __builtin_amdgcn_wmma_f32_16x16x32_bf16(false, a, false, b, (short)0, c,
                                                 false, false);
}

// CDNA5 async gather: per-lane 16B global -> LDS, tracked by ASYNCcnt.
__device__ __forceinline__ unsigned lds_off(void* p) {
  return (unsigned)(unsigned long long)(as3char*)p;
}
__device__ __forceinline__ void async_gather_b128(unsigned loff, const void* g) {
  asm volatile("global_load_async_to_lds_b128 %0, %1, off"
               :: "v"(loff), "v"(g) : "memory");
}
__device__ __forceinline__ void wait_async0() {
  asm volatile("s_wait_asynccnt 0x0" ::: "memory");
}

// ---------------------------------------------------------------------------
// prep: convert/pad x and all weights to bf16 B-column-major panels.
//   xb   [NN][160]            xb[n][k] = x[n][k] (k<130)
//   w1c  [128][160]           w1c[n][k] = W1[k][n]
//   w2c  [128][128]           w2c[n][k] = W2[k][n]
//   qkvc [12][160][160]       qkvc[m][n][k] = qkv_w[m][n][k]  (already [e][d])
//   outc [4][160][160]        outc[m][n][k] = out_w[m][n][k]  (already [o][e])
// ---------------------------------------------------------------------------
__global__ void prep_kernel(const float* __restrict__ x, const float* __restrict__ W1,
                            const float* __restrict__ W2, const float* __restrict__ qkvw,
                            const float* __restrict__ outw,
                            __bf16* __restrict__ xb, __bf16* __restrict__ w1c,
                            __bf16* __restrict__ w2c, __bf16* __restrict__ qkvc,
                            __bf16* __restrict__ outc) {
  const int XB = NN * 160, W1C = 128 * 160, W2C = 128 * 128;
  const int QKV = 12 * 160 * 160, OUT = 4 * 160 * 160;
  const int TOTAL = XB + W1C + W2C + QKV + OUT;
  for (int i = blockIdx.x * blockDim.x + threadIdx.x; i < TOTAL;
       i += gridDim.x * blockDim.x) {
    int j = i;
    if (j < XB) {
      int n = j / 160, k = j - n * 160;
      xb[j] = (k < 130) ? (__bf16)x[n * 130 + k] : (__bf16)0.0f;
      continue;
    }
    j -= XB;
    if (j < W1C) {
      int n = j / 160, k = j - n * 160;
      w1c[j] = (k < 130) ? (__bf16)W1[k * 128 + n] : (__bf16)0.0f;
      continue;
    }
    j -= W1C;
    if (j < W2C) {
      int n = j >> 7, k = j & 127;
      w2c[j] = (__bf16)W2[k * 128 + n];
      continue;
    }
    j -= W2C;
    if (j < QKV) {
      int mat = j / 25600, r = j - mat * 25600;
      int n = r / 160, k = r - n * 160;
      qkvc[j] = (n < 137 && k < 137) ? (__bf16)qkvw[mat * 18769 + n * 137 + k]
                                     : (__bf16)0.0f;
      continue;
    }
    j -= QKV;
    {
      int mat = j / 25600, r = j - mat * 25600;
      int n = r / 160, k = r - n * 160;
      outc[j] = (n < 137 && k < 137) ? (__bf16)outw[mat * 18769 + n * 137 + k]
                                     : (__bf16)0.0f;
    }
  }
}

// ---------------------------------------------------------------------------
// Dense GEMM: out[NN][128] (bf16) = A[NN][ldk] (bf16) @ Wc  (Wc is [n][k]).
// One wave per 16x16 output tile.
// ---------------------------------------------------------------------------
__global__ __launch_bounds__(128) void gemm_bf16_kernel(
    const __bf16* __restrict__ A, const __bf16* __restrict__ Wc,
    __bf16* __restrict__ out, int ldk, int ksteps) {
  int wave = (blockIdx.x * 128 + threadIdx.x) >> 5;
  int lane = threadIdx.x & 31;
  int h = lane >> 4, l16 = lane & 15;
  int mt = wave >> 3, nt = wave & 7;
  int e = nt * 16 + l16;
  v8f acc;
#pragma unroll
  for (int i = 0; i < 8; ++i) acc[i] = 0.0f;
  for (int ks = 0; ks < ksteps; ++ks) {
    int kb = ks * 32;
    acc = wmma_bf16(load_a_frag(A, ldk, mt * 16 + l16, kb, h),
                    load_b_frag(Wc, ldk, e, kb, h), acc);
  }
#pragma unroll
  for (int i = 0; i < 8; ++i)
    out[(mt * 16 + i + 8 * h) * 128 + e] = (__bf16)acc[i];
}

// ---------------------------------------------------------------------------
// Fused per-node dual-MHA attention conv + max-pool.  One wave per node,
// 4 nodes per 128-thread block.  All per-node matrices live in LDS.
// ---------------------------------------------------------------------------
__global__ __launch_bounds__(128) void attn_conv_kernel(
    const __bf16* __restrict__ hin, const __bf16* __restrict__ qkvc,
    const float* __restrict__ qkv_b, const __bf16* __restrict__ outc,
    const float* __restrict__ out_b, const float* __restrict__ blayer,
    const int* __restrict__ src, float* __restrict__ uout, int layer) {
  __shared__ __align__(32) __bf16 s_d[4][16 * 160];
  __shared__ __align__(32) __bf16 s_q[4][16 * 160];
  __shared__ __align__(32) __bf16 s_k[4][16 * 160];   // reused as o1
  __shared__ __align__(32) __bf16 s_vT[4][160 * 32];  // [e][kk], kk padded to 32
  __shared__ __align__(32) __bf16 s_o0[4][16 * 160];
  __shared__ __align__(32) __bf16 s_a[4][16 * 32];    // [q][kk]

  const int lane = threadIdx.x & 31;
  const int w = threadIdx.x >> 5;
  const int node = blockIdx.x * 4 + w;
  const int h = lane >> 4, l16 = lane & 15;

  __bf16* d = s_d[w];
  __bf16* q = s_q[w];
  __bf16* k = s_k[w];
  __bf16* vT = s_vT[w];
  __bf16* o0 = s_o0[w];
  __bf16* o1 = s_k[w];
  __bf16* a = s_a[w];

  v8bf z8;
#pragma unroll
  for (int i = 0; i < 8; ++i) z8[i] = (__bf16)0.0f;

  // ---- async gather rows 0..8, cols 0..127 of d straight into LDS --------
  // (ASYNCcnt pipe; disjoint from the DS-pipe zero-fill below, so no
  //  DS-vs-async ordering hazard exists.)
  for (int t = lane; t < 9 * 16; t += 32) {
    int s = t >> 4, c8 = t & 15;
    int nb = (s < 8) ? src[node * 8 + s] : node;
    async_gather_b128(lds_off(d + s * 160 + c8 * 8), hin + nb * 128 + c8 * 8);
  }
  // ---- zero-fill only the non-gathered regions (DS pipe) -----------------
  for (int t = lane; t < 9 * 4; t += 32) {        // rows 0..8, cols 128..159
    int s = t >> 2, c8 = (t & 3) + 16;
    *(v8bf*)(d + s * 160 + c8 * 8) = z8;
  }
  for (int t = lane; t < 7 * 20; t += 32) {       // rows 9..15, all cols
    int s = 9 + t / 20, c8 = t % 20;
    *(v8bf*)(d + s * 160 + c8 * 8) = z8;
  }
  for (int t = lane; t < 160; t += 32) {          // zero kk=16..31 of vT
    *(v8bf*)(vT + t * 32 + 16) = z8;
    *(v8bf*)(vT + t * 32 + 24) = z8;
  }
  if (lane < 9) d[lane * 160 + 128 + lane] = (__bf16)1.0f;  // pos encoding I9
  wait_async0();
  __builtin_amdgcn_fence(__ATOMIC_ACQ_REL, "wavefront");

  const float inv_sqrt = 0.08543577f;  // 1/sqrt(137)

  for (int m = 0; m < 2; ++m) {
    int base = (layer * 2 + m) * 3;
    // ---- q, k, v projections: (16x160) @ (160x160) + bias ----
    for (int j = 0; j < 3; ++j) {
      const __bf16* W = qkvc + (base + j) * 25600;
      const float* bb = qkv_b + (base + j) * 137;
      __builtin_prefetch(W + lane * 128, 0, 1);   // global_prefetch: warm L0/L2
      for (int nt = 0; nt < 10; ++nt) {
        int e = nt * 16 + l16;
        float bv = (e < 137) ? bb[e] : 0.0f;
        v8f acc;
#pragma unroll
        for (int i = 0; i < 8; ++i) acc[i] = bv;
#pragma unroll
        for (int kb = 0; kb < 160; kb += 32)
          acc = wmma_bf16(load_a_frag(d, 160, l16, kb, h),
                          load_b_frag(W, 160, e, kb, h), acc);
        if (j == 2) {  // store v transposed: vT[e][kk]
          v8bf t;
#pragma unroll
          for (int i = 0; i < 8; ++i) t[i] = (__bf16)acc[i];
          *(v8bf*)(vT + e * 32 + 8 * h) = t;
        } else {
          __bf16* dst = (j == 0) ? q : k;
#pragma unroll
          for (int i = 0; i < 8; ++i) dst[(i + 8 * h) * 160 + e] = (__bf16)acc[i];
        }
      }
    }
    // ---- scoresT = k @ q^T  (D[kk][qq]) ----
    v8f sc;
#pragma unroll
    for (int i = 0; i < 8; ++i) sc[i] = 0.0f;
#pragma unroll
    for (int kb = 0; kb < 160; kb += 32)
      sc = wmma_bf16(load_a_frag(k, 160, l16, kb, h),
                     load_b_frag(q, 160, l16, kb, h), sc);
    // ---- softmax over kk (0..8 valid); within-lane + lane^16 exchange ----
    float mx = -1e30f;
#pragma unroll
    for (int i = 0; i < 8; ++i) {
      int kk = i + 8 * h;
      float v = sc[i] * inv_sqrt;
      sc[i] = v;
      if (kk < 9 && v > mx) mx = v;
    }
    mx = fmaxf(mx, __shfl_xor(mx, 16));
    float sum = 0.0f;
#pragma unroll
    for (int i = 0; i < 8; ++i) {
      int kk = i + 8 * h;
      float ev = (kk < 9) ? __expf(sc[i] - mx) : 0.0f;
      sc[i] = ev;
      sum += ev;
    }
    sum += __shfl_xor(sum, 16);
    float inv_s = 1.0f / sum;
    v8bf av;
#pragma unroll
    for (int i = 0; i < 8; ++i) av[i] = (__bf16)(sc[i] * inv_s);
    *(v8bf*)(a + l16 * 32 + 8 * h) = av;         // a[q][kk]
    *(v8bf*)(a + l16 * 32 + 16 + 8 * h) = z8;    // pad kk 16..31
    __builtin_amdgcn_fence(__ATOMIC_ACQ_REL, "wavefront");
    // ---- o = a @ v : (16x32) @ (32x160) ----
    __bf16* ob = (m == 0) ? o0 : o1;
    for (int nt = 0; nt < 10; ++nt) {
      int e = nt * 16 + l16;
      v8f acc;
#pragma unroll
      for (int i = 0; i < 8; ++i) acc[i] = 0.0f;
      acc = wmma_bf16(load_a_frag(a, 32, l16, 0, h),
                      load_b_frag(vT, 32, e, 0, h), acc);
#pragma unroll
      for (int i = 0; i < 8; ++i) ob[(i + 8 * h) * 160 + e] = (__bf16)acc[i];
    }
  }

  // ---- y = d + o0@Wo0^T + b0 + o1@Wo1^T + b1 ; max over seq; + blayer ----
  const __bf16* Wo0 = outc + (layer * 2 + 0) * 25600;
  const __bf16* Wo1 = outc + (layer * 2 + 1) * 25600;
  const float* bo0 = out_b + (layer * 2 + 0) * 137;
  const float* bo1 = out_b + (layer * 2 + 1) * 137;
  __builtin_prefetch(Wo0 + lane * 128, 0, 1);
  __builtin_prefetch(Wo1 + lane * 128, 0, 1);
  for (int nt = 0; nt < 8; ++nt) {  // only cols < 128 survive the max-pool
    int e = nt * 16 + l16;
    float bv = bo0[e] + bo1[e];
    v8f acc;
#pragma unroll
    for (int i = 0; i < 8; ++i) acc[i] = bv + (float)d[(i + 8 * h) * 160 + e];
#pragma unroll
    for (int kb = 0; kb < 160; kb += 32)
      acc = wmma_bf16(load_a_frag(o0, 160, l16, kb, h),
                      load_b_frag(Wo0, 160, e, kb, h), acc);
#pragma unroll
    for (int kb = 0; kb < 160; kb += 32)
      acc = wmma_bf16(load_a_frag(o1, 160, l16, kb, h),
                      load_b_frag(Wo1, 160, e, kb, h), acc);
    float mm = -1e30f;
#pragma unroll
    for (int i = 0; i < 8; ++i) {
      int s = i + 8 * h;
      if (s < 9) mm = fmaxf(mm, acc[i]);
    }
    mm = fmaxf(mm, __shfl_xor(mm, 16));
    if (h == 0) uout[node * 128 + e] = mm + blayer[e];
  }
}

// ---------------------------------------------------------------------------
// PairNorm pieces (deterministic tree reductions) + apply / final kernels.
// ---------------------------------------------------------------------------
__global__ void col_mean_kernel(const float* __restrict__ u, float* __restrict__ mean) {
  __shared__ float red[256];
  int c = blockIdx.x;
  float s = 0.0f;
  for (int r = threadIdx.x; r < NN; r += 256) s += u[r * 128 + c];
  red[threadIdx.x] = s;
  __syncthreads();
  for (int off = 128; off > 0; off >>= 1) {
    if (threadIdx.x < off) red[threadIdx.x] += red[threadIdx.x + off];
    __syncthreads();
  }
  if (threadIdx.x == 0) mean[c] = red[0] / (float)NN;
}

__global__ void sq_partial_kernel(const float* __restrict__ u,
                                  const float* __restrict__ mean,
                                  float* __restrict__ part) {
  __shared__ float red[256];
  int r = blockIdx.x * 256 + threadIdx.x;
  float s = 0.0f;
  for (int c = 0; c < 128; ++c) {
    float v = u[r * 128 + c] - mean[c];
    s += v * v;
  }
  red[threadIdx.x] = s;
  __syncthreads();
  for (int off = 128; off > 0; off >>= 1) {
    if (threadIdx.x < off) red[threadIdx.x] += red[threadIdx.x + off];
    __syncthreads();
  }
  if (threadIdx.x == 0) part[blockIdx.x] = red[0];
}

__global__ void sq_final_kernel(const float* __restrict__ part, float* __restrict__ invd) {
  __shared__ float red[128];
  red[threadIdx.x] = part[threadIdx.x];
  __syncthreads();
  for (int off = 64; off > 0; off >>= 1) {
    if (threadIdx.x < off) red[threadIdx.x] += red[threadIdx.x + off];
    __syncthreads();
  }
  if (threadIdx.x == 0) invd[0] = rsqrtf(1e-6f + red[0] / (float)NN);
}

__global__ void pn_apply_kernel(const float* __restrict__ u, const float* __restrict__ mean,
                                const float* __restrict__ invd, __bf16* __restrict__ hb) {
  int i = blockIdx.x * 256 + threadIdx.x;
  int c = i & 127;
  float v = (u[i] - mean[c]) * invd[0];
  hb[i] = (__bf16)fmaxf(v, 0.0f);
}

__global__ void final_kernel(const float* __restrict__ x, const float* __restrict__ u,
                             const float* __restrict__ mean, const float* __restrict__ invd,
                             float* __restrict__ out) {
  int i = blockIdx.x * 256 + threadIdx.x;
  int c = i & 127;
  int n = i >> 7;
  float v = x[n * 130 + c] + fmaxf((u[i] - mean[c]) * invd[0], 0.0f);
  if (c == 127 && (v > 1.0f || v < -1.0f)) v = 0.0f;
  out[i] = 0.5f * v;
}

// ---------------------------------------------------------------------------
extern "C" void kernel_launch(void* const* d_in, const int* in_sizes, int n_in,
                              void* d_out, int out_size, void* d_ws, size_t ws_size,
                              hipStream_t stream) {
  const float* x = (const float*)d_in[0];
  const int* edge = (const int*)d_in[1];   // src = first NN*DEG ints
  const float* W1 = (const float*)d_in[2];
  const float* b1 = (const float*)d_in[3];
  const float* W2 = (const float*)d_in[4];
  const float* b2 = (const float*)d_in[5];
  const float* qkvw = (const float*)d_in[6];
  const float* qkvb = (const float*)d_in[7];
  const float* outw = (const float*)d_in[8];
  const float* outb = (const float*)d_in[9];

  char* ws = (char*)d_ws;
  const size_t OFF_XB = 0;                         // NN*160*2 = 10485760
  const size_t OFF_W1C = 10485760;                 // 128*160*2 = 40960
  const size_t OFF_W2C = 10526720;                 // 128*128*2 = 32768
  const size_t OFF_QKVC = 10559488;                // 12*160*160*2 = 614400
  const size_t OFF_OUTC = 11173888;                // 4*160*160*2 = 204800
  const size_t OFF_HA = 11378688;                  // NN*128*2 = 8388608
  const size_t OFF_HB = 19767296;                  // NN*128*2
  const size_t OFF_U = 28155904;                   // NN*128*4 = 16777216
  const size_t OFF_STATS = 44933120;               // small

  __bf16* xb = (__bf16*)(ws + OFF_XB);
  __bf16* w1c = (__bf16*)(ws + OFF_W1C);
  __bf16* w2c = (__bf16*)(ws + OFF_W2C);
  __bf16* qkvc = (__bf16*)(ws + OFF_QKVC);
  __bf16* outc = (__bf16*)(ws + OFF_OUTC);
  __bf16* hA = (__bf16*)(ws + OFF_HA);
  __bf16* hB = (__bf16*)(ws + OFF_HB);
  float* U = (float*)(ws + OFF_U);
  float* stats = (float*)(ws + OFF_STATS);
  float* mean = stats;         // 128
  float* invd = stats + 128;   // 1
  float* part = stats + 160;   // 128

  float* out = (float*)d_out;

  // 1) bf16/pad conversion of x and all weight panels
  {
    int total = NN * 160 + 128 * 160 + 128 * 128 + 12 * 25600 + 4 * 25600;
    prep_kernel<<<(total + 255) / 256, 256, 0, stream>>>(x, W1, W2, qkvw, outw,
                                                         xb, w1c, w2c, qkvc, outc);
  }
  // 2) h1 = x @ W1   (K padded 130->160)
  gemm_bf16_kernel<<<4096, 128, 0, stream>>>(xb, w1c, hA, 160, 5);
  // 3) layer-0 attention conv -> U
  attn_conv_kernel<<<NN / 4, 128, 0, stream>>>(hA, qkvc, qkvb, outc, outb, b1,
                                               edge, U, 0);
  // 4) pairnorm + relu -> hB (bf16)
  col_mean_kernel<<<128, 256, 0, stream>>>(U, mean);
  sq_partial_kernel<<<128, 256, 0, stream>>>(U, mean, part);
  sq_final_kernel<<<1, 128, 0, stream>>>(part, invd);
  pn_apply_kernel<<<NN * 128 / 256, 256, 0, stream>>>(U, mean, invd, hB);
  // 5) h2 = hB @ W2
  gemm_bf16_kernel<<<4096, 128, 0, stream>>>(hB, w2c, hA, 128, 4);
  // 6) layer-1 attention conv -> U
  attn_conv_kernel<<<NN / 4, 128, 0, stream>>>(hA, qkvc, qkvb, outc, outb, b2,
                                               edge, U, 1);
  // 7) pairnorm stats for layer 2
  col_mean_kernel<<<128, 256, 0, stream>>>(U, mean);
  sq_partial_kernel<<<128, 256, 0, stream>>>(U, mean, part);
  sq_final_kernel<<<1, 128, 0, stream>>>(part, invd);
  // 8) fused pairnorm+relu + residual + gate + /2
  final_kernel<<<NN * 128 / 256, 256, 0, stream>>>(x, U, mean, invd, out);
}
